// MultiheadAttention_60868276519275
// MI455X (gfx1250) — matrix operand order repaired and checked by reading, not compiled, not run on hardware
//
#include <hip/hip_runtime.h>
#include <hip/hip_bf16.h>

// ---------------------------------------------------------------------------
// MultiheadAttention for MI455X (gfx1250), wave32 + WMMA bf16 + async LDS.
//   B=2, N=S=2048, DIM=1024, H=16, hd=64, SCALE=1/8.
// Pipeline:
//   1) transpose+cvt W_qkv, W_out -> bf16 W^T  (B-fragment friendly layout)
//   2) gemm (x @ Wq +b) -> Q  [B,H,N,64]  bf16
//      gemm (x @ Wk +b) -> K  [B,H,S,64]  bf16
//      gemm (x @ Wv +b) -> Vt [B,H,64,S]  bf16 (transposed for P@V B-frags)
//   3) flash attention (K/V double-buffered via GLOBAL_LOAD_ASYNC_TO_LDS)
//      -> ctx [B,N,1024] bf16
//   4) gemm (ctx @ Wout + b) -> d_out f32
// ---------------------------------------------------------------------------

typedef __attribute__((ext_vector_type(8)))  float        v8f;
typedef __attribute__((ext_vector_type(8)))  unsigned int v8u;
typedef __attribute__((ext_vector_type(16))) __bf16       v16bf;

__device__ __forceinline__ unsigned short f2bf(float f) {
  unsigned int u = __float_as_uint(f);
  return (unsigned short)((u + 0x7FFFu + ((u >> 16) & 1u)) >> 16);
}
__device__ __forceinline__ unsigned int pack2(unsigned short lo, unsigned short hi) {
  return (unsigned int)lo | ((unsigned int)hi << 16);
}

// Wave-relative LDS byte address (VDST operand of async-to-LDS ops).
__device__ __forceinline__ unsigned lds_off(const void* p) {
  return (unsigned)(unsigned long long)(const __attribute__((address_space(3))) void*)p;
}

// Async 16-byte global->LDS copy, tracked by ASYNCcnt.
__device__ __forceinline__ void async_cp16(unsigned lds_addr, const void* gp) {
  asm volatile("global_load_async_to_lds_b128 %0, %1, off"
               :: "v"(lds_addr), "v"((unsigned long long)gp)
               : "memory");
}

// A-matrix 16x32 bf16 fragment (ISA 7.12.2): lane<16 row M=lane, K pairs
// {0..7,16..23}; lane>=16 same M, K pairs {8..15,24..31}. rowk = &A[row][k0].
__device__ __forceinline__ v16bf load_a_bf16(const unsigned short* rowk, int hf) {
  v8u a;
#pragma unroll
  for (int j = 0; j < 4; ++j) {
    a[j]     = *(const unsigned int*)(rowk + hf * 8 + 2 * j);
    a[4 + j] = *(const unsigned int*)(rowk + 16 + hf * 8 + 2 * j);
  }
  return __builtin_bit_cast(v16bf, a);
}
// Same fragment but converting f32 source on the fly (QKV projection inputs).
__device__ __forceinline__ v16bf load_a_f32(const float* rowk, int hf) {
  v8u a;
#pragma unroll
  for (int j = 0; j < 4; ++j) {
    const float* p0 = rowk + hf * 8 + 2 * j;
    const float* p1 = rowk + 16 + hf * 8 + 2 * j;
    a[j]     = pack2(f2bf(p0[0]), f2bf(p0[1]));
    a[4 + j] = pack2(f2bf(p1[0]), f2bf(p1[1]));
  }
  return __builtin_bit_cast(v16bf, a);
}
__device__ __forceinline__ v16bf load_a(const float* p, int hf)          { return load_a_f32(p, hf); }
__device__ __forceinline__ v16bf load_a(const unsigned short* p, int hf) { return load_a_bf16(p, hf); }

// B-matrix 32x16 bf16 fragment: lane = column N, same K order as A
// (lanes 0-15: K {0..7,16..23}, lanes 16-31: K {8..15,24..31}). rowk = &(B^T)[col][k0].
__device__ __forceinline__ v16bf load_b_bf16(const unsigned short* rowk, int hf) {
  v8u b;
#pragma unroll
  for (int j = 0; j < 4; ++j) {
    b[j]     = *(const unsigned int*)(rowk + hf * 8 + 2 * j);
    b[4 + j] = *(const unsigned int*)(rowk + 16 + hf * 8 + 2 * j);
  }
  return __builtin_bit_cast(v16bf, b);
}

__device__ __forceinline__ v8f wmma_bf16(v16bf a, v16bf b, v8f c) {
  return __builtin_amdgcn_wmma_f32_16x16x32_bf16(false, a, false, b, (short)0, c,
                                                 false, false);
}

// --------------------------- weight transpose+cvt ---------------------------
// Wt[n][k] = bf16(W[k][n]); W is [K][N] row-major f32.
__global__ __launch_bounds__(256) void tr_cvt_k(const float* __restrict__ W,
                                                unsigned short* __restrict__ Wt,
                                                int K, int N) {
  __shared__ float tile[32][33];
  int tx = threadIdx.x, ty = threadIdx.y;          // block 32x8
  int kb = blockIdx.y * 32, nb = blockIdx.x * 32;
#pragma unroll
  for (int i = ty; i < 32; i += 8)
    tile[i][tx] = W[(size_t)(kb + i) * N + nb + tx];
  __syncthreads();
#pragma unroll
  for (int i = ty; i < 32; i += 8)
    Wt[(size_t)(nb + i) * K + kb + tx] = f2bf(tile[tx][i]);
}

// ------------------------------- tiled GEMM ---------------------------------
// C[4096][1024] = A[4096][1024] @ Bt^T + bias.  Block = 8 waves -> 64M x 128N
// tile, each wave 32x32 (2x2 WMMA tiles), K chunked by 32.
// MODE 0: f32 row-major out (+bias)            -> final projection
// MODE 1: bf16 out, layout [B][H][L][64]       -> Q / K buffers
// MODE 2: bf16 out, layout [B][H][64][S]       -> V transposed buffer
template <typename AT, int MODE>
__global__ __launch_bounds__(256) void gemm_k(const AT* __restrict__ A,
                                              const unsigned short* __restrict__ Bt,
                                              const float* __restrict__ bias,
                                              void* __restrict__ outp) {
  constexpr int K = 1024;
  int wave = threadIdx.x >> 5, lane = threadIdx.x & 31;
  int hf = lane >> 4, lm = lane & 15;
  int m0 = blockIdx.y * 64 + (wave >> 2) * 32;
  int n0 = blockIdx.x * 128 + (wave & 3) * 32;

  v8f acc[2][2] = {};
  const AT* arow0 = A + (size_t)(m0 + lm) * K;
  const AT* arow1 = arow0 + (size_t)16 * K;
  const unsigned short* brow0 = Bt + (size_t)(n0 + lm) * K;
  const unsigned short* brow1 = brow0 + (size_t)16 * K;

  for (int k0 = 0; k0 < K; k0 += 32) {
    __builtin_prefetch((const void*)(brow0 + k0 + 256), 0, 1);
    __builtin_prefetch((const void*)(brow1 + k0 + 256), 0, 1);
    v16bf a0 = load_a(arow0 + k0, hf);
    v16bf a1 = load_a(arow1 + k0, hf);
    v16bf b0 = load_b_bf16(brow0 + k0, hf);
    v16bf b1 = load_b_bf16(brow1 + k0, hf);
    acc[0][0] = wmma_bf16(a0, b0, acc[0][0]);
    acc[0][1] = wmma_bf16(a0, b1, acc[0][1]);
    acc[1][0] = wmma_bf16(a1, b0, acc[1][0]);
    acc[1][1] = wmma_bf16(a1, b1, acc[1][1]);
  }

  // C/D layout: lane column N = lm; VGPR r holds row M = r + 8*half.
#pragma unroll
  for (int i = 0; i < 2; ++i)
#pragma unroll
    for (int j = 0; j < 2; ++j)
#pragma unroll
      for (int r = 0; r < 8; ++r) {
        int Mg = m0 + i * 16 + r + 8 * hf;
        int c  = n0 + j * 16 + lm;
        float val = acc[i][j][r] + bias[c];
        if constexpr (MODE == 0) {
          ((float*)outp)[(size_t)Mg * 1024 + c] = val;
        } else if constexpr (MODE == 1) {
          int b = Mg >> 11, n = Mg & 2047, h = c >> 6, d = c & 63;
          ((unsigned short*)outp)[(((size_t)(b * 16 + h)) * 2048 + n) * 64 + d] =
              f2bf(val);
        } else {
          int b = Mg >> 11, n = Mg & 2047, h = c >> 6, d = c & 63;
          ((unsigned short*)outp)[(((size_t)(b * 16 + h)) * 64 + d) * 2048 + n] =
              f2bf(val);
        }
      }
}

// ---------------------------- flash attention -------------------------------
// Block = 4 waves, each wave owns 16 query rows of one (b,h). Streams S in
// chunks of 32 with online softmax. K/V chunks are staged into double-buffered
// LDS tiles with GLOBAL_LOAD_ASYNC_TO_LDS_B128 (ASYNCcnt) so the copy of
// chunk i+1 overlaps the WMMA/softmax work on chunk i.
__global__ __launch_bounds__(128) void attn_k(const unsigned short* __restrict__ Q,
                                              const unsigned short* __restrict__ Kb,
                                              const unsigned short* __restrict__ Vt,
                                              unsigned short* __restrict__ ctx) {
  constexpr float SCALE = 0.125f;  // 64^-0.5
  __shared__ unsigned short ldsK[2][32][64];   // [buf][s][hd]   4KB per buf
  __shared__ unsigned short ldsV[2][64][32];   // [buf][hd][s]   4KB per buf
  __shared__ unsigned short pT[4][16][32];     // per-wave P re-layout tile

  int wave = threadIdx.x >> 5, lane = threadIdx.x & 31;
  int hf = lane >> 4, lm = lane & 15;
  int bh = blockIdx.y;                         // b*16 + h
  int n0 = blockIdx.x * 64 + wave * 16;

  const unsigned short* qrow = Q + ((size_t)bh * 2048 + n0 + lm) * 64;
  v16bf aq0 = load_a_bf16(qrow, hf);           // K = 0..31  of head dim
  v16bf aq1 = load_a_bf16(qrow + 32, hf);      // K = 32..63

  // Stage one 32-wide S chunk (K: 32x64 bf16, V: 64x32 bf16) into LDS buffer.
  // 128 threads x 4 async b128 ops = 8KB; each wave issues exactly 4 asyncs.
  auto stage = [&](int nbuf, int sN) {
    unsigned t = threadIdx.x;
#pragma unroll
    for (int i = 0; i < 2; ++i) {              // K tile: 256 x 16B chunks
      unsigned c = t + i * 128;
      async_cp16(lds_off(&ldsK[nbuf][c >> 3][(c & 7) * 8]),
                 Kb + ((size_t)bh * 2048 + sN + (c >> 3)) * 64 + (c & 7) * 8);
    }
#pragma unroll
    for (int i = 0; i < 2; ++i) {              // V tile: 256 x 16B chunks
      unsigned c = t + i * 128;
      async_cp16(lds_off(&ldsV[nbuf][c >> 2][(c & 3) * 8]),
                 Vt + ((size_t)bh * 64 + (c >> 2)) * 2048 + sN + (c & 3) * 8);
    }
  };

  stage(0, 0);                                 // prologue: chunk 0 -> buf 0

  float mrun[8], lrun[8];
#pragma unroll
  for (int r = 0; r < 8; ++r) { mrun[r] = -1e30f; lrun[r] = 0.f; }
  v8f o[4] = {};

  for (int s0 = 0; s0 < 2048; s0 += 32) {
    int buf = (s0 >> 5) & 1;
    // Issue next chunk (wraps to 0 on the last iteration; never read then).
    stage(buf ^ 1, (s0 + 32) & 2047);
    // 8 asyncs in flight per wave; <=4 left guarantees current buf landed
    // (async loads complete in order).
    asm volatile("s_wait_asynccnt 0x4" ::: "memory");
    __syncthreads();                           // all waves' copies visible

    // ---- scores: Q(16x64) @ K^T -> two 16x16 tiles (cols s0..15, s16..31)
    const unsigned short* kb0 = &ldsK[buf][lm][0];
    const unsigned short* kb1 = &ldsK[buf][16 + lm][0];
    v8f c0 = {}, c1 = {};
    c0 = wmma_bf16(aq0, load_b_bf16(kb0, hf), c0);
    c0 = wmma_bf16(aq1, load_b_bf16(kb0 + 32, hf), c0);
    c1 = wmma_bf16(aq0, load_b_bf16(kb1, hf), c1);
    c1 = wmma_bf16(aq1, load_b_bf16(kb1 + 32, hf), c1);

    // ---- online softmax (row data lives in one 16-lane half-wave)
    float p0[8], p1[8];
#pragma unroll
    for (int r = 0; r < 8; ++r) {
      float v0 = c0[r] * SCALE, v1 = c1[r] * SCALE;
      float mx = fmaxf(v0, v1);
#pragma unroll
      for (int off = 1; off < 16; off <<= 1) mx = fmaxf(mx, __shfl_xor(mx, off, 16));
      float mn = fmaxf(mrun[r], mx);
      float corr = __expf(mrun[r] - mn);
      p0[r] = __expf(v0 - mn);
      p1[r] = __expf(v1 - mn);
      float sm = p0[r] + p1[r];
#pragma unroll
      for (int off = 1; off < 16; off <<= 1) sm += __shfl_xor(sm, off, 16);
      lrun[r] = lrun[r] * corr + sm;
      mrun[r] = mn;
#pragma unroll
      for (int t = 0; t < 4; ++t) o[t][r] *= corr;
    }

    // ---- re-layout P (C-layout) -> A-fragment via per-wave LDS tile
#pragma unroll
    for (int r = 0; r < 8; ++r) {
      int M = r + 8 * hf;
      pT[wave][M][lm]      = f2bf(p0[r]);
      pT[wave][M][16 + lm] = f2bf(p1[r]);
    }
    __syncthreads();
    v16bf pa = load_a_bf16(&pT[wave][lm][0], hf);

    // ---- O += P(16x32) @ V(32x64): B-frags from LDS V tile [hd][s]
#pragma unroll
    for (int t = 0; t < 4; ++t)
      o[t] = wmma_bf16(pa, load_b_bf16(&ldsV[buf][t * 16 + lm][0], hf), o[t]);

    __syncthreads();  // reads of buf done before next iter overwrites it
  }

  // ---- normalize and store ctx [b][n][h*64+d] bf16
  int b = bh >> 4, h = bh & 15;
  float inv[8];
#pragma unroll
  for (int r = 0; r < 8; ++r) inv[r] = 1.f / lrun[r];
#pragma unroll
  for (int t = 0; t < 4; ++t)
#pragma unroll
    for (int r = 0; r < 8; ++r) {
      size_t row = (size_t)b * 2048 + n0 + r + 8 * hf;
      size_t col = (size_t)h * 64 + t * 16 + lm;
      ctx[row * 1024 + col] = f2bf(o[t][r] * inv[r]);
    }
}

// ------------------------------- launcher -----------------------------------
extern "C" void kernel_launch(void* const* d_in, const int* in_sizes, int n_in,
                              void* d_out, int out_size, void* d_ws, size_t ws_size,
                              hipStream_t stream) {
  (void)in_sizes; (void)n_in; (void)out_size; (void)ws_size;
  const float* q    = (const float*)d_in[0];
  const float* kk   = (const float*)d_in[1];
  const float* vv   = (const float*)d_in[2];
  const float* Wqkv = (const float*)d_in[3];
  const float* bqkv = (const float*)d_in[4];
  const float* Wout = (const float*)d_in[5];
  const float* bout = (const float*)d_in[6];

  // workspace layout (bf16), ~40 MB total
  char* ws = (char*)d_ws;
  unsigned short* WtQKV = (unsigned short*)ws;                       // [3072][1024]
  unsigned short* WtOut = (unsigned short*)(ws + 3072ull * 1024 * 2);// [1024][1024]
  unsigned short* Qb    = (unsigned short*)(ws + (3072ull + 1024) * 1024 * 2);
  unsigned short* Kbf   = Qb  + 2ull * 16 * 2048 * 64;
  unsigned short* Vtb   = Kbf + 2ull * 16 * 2048 * 64;
  unsigned short* ctx   = Vtb + 2ull * 16 * 2048 * 64;

  dim3 tb(32, 8);
  tr_cvt_k<<<dim3(96, 32), tb, 0, stream>>>(Wqkv, WtQKV, 1024, 3072);
  tr_cvt_k<<<dim3(32, 32), tb, 0, stream>>>(Wout, WtOut, 1024, 1024);

  dim3 gg(8, 64), gb(256);
  gemm_k<float, 1><<<gg, gb, 0, stream>>>(q,  WtQKV,                    bqkv,        (void*)Qb);
  gemm_k<float, 1><<<gg, gb, 0, stream>>>(kk, WtQKV + 1024ull * 1024,   bqkv + 1024, (void*)Kbf);
  gemm_k<float, 2><<<gg, gb, 0, stream>>>(vv, WtQKV + 2048ull * 1024,   bqkv + 2048, (void*)Vtb);

  attn_k<<<dim3(32, 32), 128, 0, stream>>>(Qb, Kbf, Vtb, ctx);

  gemm_k<unsigned short, 0><<<gg, gb, 0, stream>>>(ctx, WtOut, bout, d_out);
}
